// GraphGPSNet_32238024524285
// MI455X (gfx1250) — matrix operand order repaired and checked by reading, not compiled, not run on hardware
//
#include <hip/hip_runtime.h>
#include <hip/hip_bf16.h>
#include <math.h>
#include <stdint.h>

#define N_NODES  50000
#define N_EDGES  600000
#define N_GRAPHS 100
#define NPG      500
#define HID      128
#define HEADS    4
#define DH       32
#define KPAD_NODE 96   // concat(x[64], pe[16]) = 80 -> pad to 96 (3x32)
#define KPAD_EDGE 32   // edge_attr 16 -> pad to 32

typedef __bf16 bf16_t;
typedef bf16_t bf16x2  __attribute__((ext_vector_type(2)));
typedef bf16_t bf16x8  __attribute__((ext_vector_type(8)));
typedef bf16_t bf16x16 __attribute__((ext_vector_type(16)));
typedef float  f32x8   __attribute__((ext_vector_type(8)));
typedef unsigned int uint32x4 __attribute__((ext_vector_type(4)));
typedef int int32x4 __attribute__((ext_vector_type(4)));
typedef int int32x8 __attribute__((ext_vector_type(8)));

__device__ __forceinline__ float gelu_erf(float x) {
    return 0.5f * x * (1.0f + erff(x * 0.70710678118654752f));
}

// ---------------------------------------------------------------------------
// Blocked GEMM: C[M,N] = act(A[M,K] @ W + bias); A fp32 row-major [M,K],
// W pre-converted bf16 *transposed* [N,K]. Block = 128 threads (4 waves) ->
// 64x64 output tile.
//  - B tile (64 N-rows x 32 K, bf16, K-contiguous) is DMA'd into LDS by the
//    Tensor Data Mover: wave 0 issues TENSOR_LOAD_TO_LDS with an in-register
//    D# (2D tile, data_size=2B, tile_dim0=32, tile_dim1=64, stride=K), then
//    s_wait_tensorcnt 0 + workgroup barrier.
//  - A tile (64x32) staged by all threads: float4 global loads, packed
//    2xbf16 ds_store_b32 writes.
//  - wave w owns N-subtile w; one B fragment per K-step reused across 4
//    M-subtiles -> 4 v_wmma_f32_16x16x32_bf16 per K-step per wave.
// Requires K%32==0, N%64==0; M tail via clamped loads / guarded stores.
// ---------------------------------------------------------------------------
__global__ void __launch_bounds__(128)
gemm_bf16_wmma(const float* __restrict__ A, const bf16_t* __restrict__ Wt,
               const float* __restrict__ bias, float* __restrict__ C,
               int M, int N, int K, int act)
{
    __shared__ __align__(16) bf16_t lAs[64][32];   // [m][k]  4 KB
    __shared__ __align__(16) bf16_t lBs[64][32];   // [n][k]  4 KB

    const int wave  = threadIdx.x >> 5;
    const int lane  = threadIdx.x & 31;
    const int half  = lane >> 4;
    const int l15   = lane & 15;
    const int tileM = blockIdx.x * 64;
    const int tileN = blockIdx.y * 64;

    f32x8 acc[4] = {};

    for (int k0 = 0; k0 < K; k0 += 32) {
        // ---- TDM: DMA 64x32 bf16 weight tile (rows tileN.., cols k0..) ----
        if (wave == 0) {
            unsigned long long ga =
                (unsigned long long)(uintptr_t)(Wt + (size_t)tileN * K + k0);
            unsigned ldsa = (unsigned)(uintptr_t)&lBs[0][0];
            uint32x4 g0;
            g0[0] = 1u;                                       // count=1 (valid)
            g0[1] = ldsa;                                     // lds_addr
            g0[2] = (unsigned)(ga & 0xffffffffu);             // global_addr lo
            g0[3] = (unsigned)((ga >> 32) & 0x01ffffffu) | (2u << 30); // hi|type=2
            int32x8 g1;
            g1[0] = (int)(1u << 16);                          // data_size=2B
            g1[1] = (int)(((unsigned)K & 0xffffu) << 16);     // tensor_dim0 lo
            g1[2] = (int)((((unsigned)K >> 16) & 0xffffu)     // tensor_dim0 hi
                          | (64u << 16));                     // tensor_dim1=64
            g1[3] = (int)(32u << 16);                         // tile_dim0=32
            g1[4] = 64;                                       // tile_dim1=64
            g1[5] = (int)K;                                   // dim0_stride=K
            g1[6] = 0;
            g1[7] = 0;
            int32x4 z4 = {};
            int32x8 z8 = {};
            __builtin_amdgcn_tensor_load_to_lds(g0, g1, z4, z4, z8, 0);
            __builtin_amdgcn_s_wait_tensorcnt(0);
        }
        // ---- stage A tile (64 rows x 32 k) as bf16, packed b32 LDS stores --
#pragma unroll
        for (int i = 0; i < 4; ++i) {
            int idx = threadIdx.x + i * 128;        // 0..511
            int r   = idx >> 3;                     // 0..63
            int c   = (idx & 7) << 2;               // 0,4,..,28
            int rowg = tileM + r;
            if (rowg > M - 1) rowg = M - 1;         // clamp tail reads
            const float4 v = *(const float4*)&A[(size_t)rowg * K + k0 + c];
            bf16x2 p0; p0[0] = (bf16_t)v.x; p0[1] = (bf16_t)v.y;
            bf16x2 p1; p1[0] = (bf16_t)v.z; p1[1] = (bf16_t)v.w;
            *(bf16x2*)&lAs[r][c]     = p0;
            *(bf16x2*)&lAs[r][c + 2] = p1;
        }
        __syncthreads();

        // ---- B fragment: lane = N col, K vals 8h..8h+7 and 16+8h..23+8h (2x ds_b128) ---
        const bf16x8 b0 = *(const bf16x8*)&lBs[wave * 16 + l15][half * 8];
        const bf16x8 b1 = *(const bf16x8*)&lBs[wave * 16 + l15][16 + half * 8];
        const bf16x16 bfrag = __builtin_shufflevector(
            b0, b1, 0, 1, 2, 3, 4, 5, 6, 7, 8, 9, 10, 11, 12, 13, 14, 15);

        // ---- 4 M-subtiles, reuse bfrag ----
#pragma unroll
        for (int t = 0; t < 4; ++t) {
            const bf16x8 a0 = *(const bf16x8*)&lAs[t * 16 + l15][half * 8];
            const bf16x8 a1 = *(const bf16x8*)&lAs[t * 16 + l15][16 + half * 8];
            const bf16x16 afrag = __builtin_shufflevector(
                a0, a1, 0, 1, 2, 3, 4, 5, 6, 7, 8, 9, 10, 11, 12, 13, 14, 15);
            acc[t] = __builtin_amdgcn_wmma_f32_16x16x32_bf16(
                false, afrag, false, bfrag, (short)0, acc[t], false, false);
        }
        __syncthreads();
    }

    // ---- epilogue: bias + activation + store ----
    const int col = tileN + wave * 16 + l15;
    const float bv = bias ? bias[col] : 0.0f;
#pragma unroll
    for (int t = 0; t < 4; ++t) {
#pragma unroll
        for (int j = 0; j < 8; ++j) {
            int row = tileM + t * 16 + (half << 3) + j;
            if (row < M) {
                float v = acc[t][j] + bv;
                if (act == 1) v = gelu_erf(v);
                C[(size_t)row * N + col] = v;
            }
        }
    }
}

// ---------------------------------------------------------------------------
// Weight convert: fp32 [K,N] -> bf16 transposed [N,KP], zero-padding K..KP.
// ---------------------------------------------------------------------------
__global__ void convert_w_kernel(const float* __restrict__ W,
                                 bf16_t* __restrict__ Wt,
                                 int K, int N, int KP)
{
    int i = blockIdx.x * blockDim.x + threadIdx.x;
    if (i >= N * KP) return;
    int n = i / KP, k = i - n * KP;
    Wt[(size_t)n * KP + k] = (k < K) ? (bf16_t)W[(size_t)k * N + n] : (bf16_t)0.0f;
}

// ---------------------------------------------------------------------------
// Strided row copy (zero-padded K packing of activations).
// ---------------------------------------------------------------------------
__global__ void copy_rows_kernel(const float* __restrict__ src, int scols,
                                 float* __restrict__ dst, int dcols,
                                 long rows, int cols)
{
    long i = (long)blockIdx.x * blockDim.x + threadIdx.x;
    if (i >= rows * (long)cols) return;
    long r = i / cols;
    int  c = (int)(i - r * cols);
    dst[r * (long)dcols + c] = src[r * (long)scols + c];
}

// ---------------------------------------------------------------------------
// GINE message + scatter-add: aggr[dst] += relu(h[src] + elin[e]).
// One block per edge, thread = feature dim. Aggr (25.6 MB) is L2-resident.
// ---------------------------------------------------------------------------
__global__ void __launch_bounds__(HID)
gine_scatter_kernel(const float* __restrict__ h, const float* __restrict__ elin,
                    const int* __restrict__ ei, float* __restrict__ aggr)
{
    int e = blockIdx.x;
    int d = threadIdx.x;
    int s = ei[e];
    int t = ei[N_EDGES + e];
    float v = h[(size_t)s * HID + d] + elin[(size_t)e * HID + d];
    v = fmaxf(v, 0.0f);
    atomicAdd(&aggr[(size_t)t * HID + d], v);
}

__global__ void gine_combine_kernel(const float* __restrict__ h,
                                    const float* __restrict__ aggr,
                                    const float* __restrict__ eps,
                                    float* __restrict__ out, int n)
{
    int i = blockIdx.x * blockDim.x + threadIdx.x;
    if (i < n) out[i] = (1.0f + eps[0]) * h[i] + aggr[i];
}

__global__ void add_kernel(const float* __restrict__ a, const float* __restrict__ b,
                           float* __restrict__ c, int n)
{
    int i = blockIdx.x * blockDim.x + threadIdx.x;
    if (i < n) c[i] = a[i] + b[i];
}

// ---------------------------------------------------------------------------
// out = LN(a + b) * gamma + beta  (b optional). One wave per 128-wide row.
// ---------------------------------------------------------------------------
__global__ void __launch_bounds__(128)
add_ln_kernel(const float* __restrict__ a, const float* __restrict__ b,
              const float* __restrict__ g, const float* __restrict__ be,
              float* __restrict__ out, int rows)
{
    int row  = blockIdx.x * 4 + (threadIdx.x >> 5);
    int lane = threadIdx.x & 31;
    if (row >= rows) return;
    float v[4];
    float s = 0.0f, ss = 0.0f;
#pragma unroll
    for (int i = 0; i < 4; ++i) {
        int c = lane + i * 32;
        float x = a[(size_t)row * HID + c];
        if (b) x += b[(size_t)row * HID + c];
        v[i] = x; s += x; ss += x * x;
    }
#pragma unroll
    for (int off = 16; off > 0; off >>= 1) {
        s  += __shfl_xor(s,  off, 32);
        ss += __shfl_xor(ss, off, 32);
    }
    const float mean = s * (1.0f / HID);
    const float var  = ss * (1.0f / HID) - mean * mean;
    const float r    = rsqrtf(var + 1e-5f);
#pragma unroll
    for (int i = 0; i < 4; ++i) {
        int c = lane + i * 32;
        out[(size_t)row * HID + c] = (v[i] - mean) * r * g[c] + be[c];
    }
}

// ---------------------------------------------------------------------------
// Per-graph dense attention (500 keys, dh=32, 4 heads). One wave per
// (node, head); lane = head-dim; online softmax; scores broadcast via
// wave32 shfl_xor butterfly.
// ---------------------------------------------------------------------------
__global__ void __launch_bounds__(128)
attn_kernel(const float* __restrict__ Q, const float* __restrict__ Kb,
            const float* __restrict__ Vb, float* __restrict__ O)
{
    int wid  = blockIdx.x * 4 + (threadIdx.x >> 5);
    int lane = threadIdx.x & 31;
    int head = wid & (HEADS - 1);
    int node = wid >> 2;
    if (node >= N_NODES) return;
    int graph = node / NPG;
    int base  = graph * NPG;
    const float scale = 0.17677669529663687f; // 1/sqrt(32)
    float qd = Q[(size_t)node * HID + head * DH + lane] * scale;
    float m = -3.0e38f, l = 0.0f, o = 0.0f;
    for (int j = 0; j < NPG; ++j) {
        size_t r = (size_t)(base + j) * HID + head * DH + lane;
        float p = qd * Kb[r];
#pragma unroll
        for (int off = 16; off > 0; off >>= 1) p += __shfl_xor(p, off, 32);
        float vd = Vb[r];
        float mn = fmaxf(m, p);
        float eo = __expf(m - mn);
        float ep = __expf(p - mn);
        l = l * eo + ep;
        o = o * eo + ep * vd;
        m = mn;
    }
    O[(size_t)node * HID + head * DH + lane] = o / l;
}

// ---------------------------------------------------------------------------
// Mean pool per graph + readout MLP (128 -> 128 GELU -> 1). grid=100, block=128.
// ---------------------------------------------------------------------------
__global__ void __launch_bounds__(HID)
pool_readout_kernel(const float* __restrict__ h, const float* __restrict__ W0,
                    const float* __restrict__ b0, const float* __restrict__ W1,
                    const float* __restrict__ b1, float* __restrict__ out)
{
    __shared__ float sm[HID];
    __shared__ float red[HID];
    int g = blockIdx.x, t = threadIdx.x;
    float s = 0.0f;
    for (int i = 0; i < NPG; ++i) s += h[(size_t)(g * NPG + i) * HID + t];
    sm[t] = s * (1.0f / NPG);
    __syncthreads();
    float acc = b0[t];
    for (int d = 0; d < HID; ++d) acc += sm[d] * W0[d * HID + t];
    red[t] = gelu_erf(acc) * W1[t];
    __syncthreads();
    for (int off = 64; off > 0; off >>= 1) {
        if (t < off) red[t] += red[t + off];
        __syncthreads();
    }
    if (t == 0) out[g] = red[0] + b1[0];
}

// ---------------------------------------------------------------------------
// Host orchestration.
// Input flattening assumption (jax pytree: dict keys sorted per level, after
// the 5 top-level arrays in setup_inputs() insertion order):
//   0:x 1:edge_index 2:batch 3:edge_attr 4:lap_pe
//   5..10  edge_mlp  (W0,B0,W1,B1,W2,B2)
//   11+27L layers[L]: bk,bo,bq,bv,wk,wo,wq,wv, gine_eps, glW,glB,
//                     gnW0,gnB0,gnW1,gnB1,gnW2,gnB2,
//                     ln1g,ln1b,ln2g,ln2b,ln3g,ln3b, mW0,mB0,mW1,mB1
//   119..124 node_mlp, 125..128 postnet, 129..132 readout
// ---------------------------------------------------------------------------
extern "C" void kernel_launch(void* const* d_in, const int* in_sizes, int n_in,
                              void* d_out, int out_size, void* d_ws, size_t ws_size,
                              hipStream_t stream)
{
    (void)in_sizes; (void)n_in; (void)out_size; (void)ws_size;
    const float* x         = (const float*)d_in[0];
    const int*   ei        = (const int*)d_in[1];
    const float* edge_attr = (const float*)d_in[3];
    const float* lap_pe    = (const float*)d_in[4];
    auto P = [&](int i) { return (const float*)d_in[i]; };

    // workspace layout (fp32 elements), ~1.05 GB total
    float* w = (float*)d_ws;
    size_t off = 0;
    auto alloc = [&](size_t n) { float* p = w + off; off += n; return p; };
    float* EA   = alloc((size_t)N_EDGES * HID);        // edge features (persist)
    float* EBUF = alloc((size_t)N_EDGES * HID);        // edge scratch / e_lin
    float* EIN  = alloc((size_t)N_EDGES * KPAD_EDGE);  // padded edge input
    float* NIN  = alloc((size_t)N_NODES * KPAD_NODE);  // padded concat input
    float* H    = alloc((size_t)N_NODES * HID);
    float* TA   = alloc((size_t)N_NODES * HID);
    float* TB   = alloc((size_t)N_NODES * HID);
    float* TC   = alloc((size_t)N_NODES * HID);
    float* AGGR = alloc((size_t)N_NODES * HID);
    float* HLOC = alloc((size_t)N_NODES * HID);
    float* HATT = alloc((size_t)N_NODES * HID);
    float* QB   = alloc((size_t)N_NODES * HID);
    float* KB   = alloc((size_t)N_NODES * HID);
    float* VB   = alloc((size_t)N_NODES * HID);
    float* AOUT = alloc((size_t)N_NODES * HID);
    float* FFNH = alloc((size_t)N_NODES * 2 * HID);
    // bf16 weight arena (~2 MB): transposed, K zero-padded
    bf16_t* warena = (bf16_t*)alloc(700000);           // 1.4 M bf16 slots
    size_t woff = 0;

    const int nEl = N_NODES * HID;

    auto convw = [&](const float* W, int K, int N, int KP) {
        bf16_t* dst = warena + woff;
        woff += (size_t)N * KP;
        int n = N * KP;
        convert_w_kernel<<<(n + 255) / 256, 256, 0, stream>>>(W, dst, K, N, KP);
        return (const bf16_t*)dst;
    };
    auto gemm = [&](const float* A, const bf16_t* Wt, const float* bias, float* C,
                    int M, int N, int K, int act) {
        dim3 grid((unsigned)((M + 63) / 64), (unsigned)(N / 64));
        gemm_bf16_wmma<<<grid, 128, 0, stream>>>(A, Wt, bias, C, M, N, K, act);
    };
    auto cpyrows = [&](const float* s, int sc, float* d, int dc, long rows, int cols) {
        long n = rows * (long)cols;
        copy_rows_kernel<<<(unsigned)((n + 255) / 256), 256, 0, stream>>>(s, sc, d, dc, rows, cols);
    };
    auto addv = [&](const float* a, const float* b, float* c) {
        add_kernel<<<(nEl + 255) / 256, 256, 0, stream>>>(a, b, c, nEl);
    };
    auto addln = [&](const float* a, const float* b, const float* g,
                     const float* be, float* o) {
        add_ln_kernel<<<(N_NODES + 3) / 4, 128, 0, stream>>>(a, b, g, be, o, N_NODES);
    };

    // zero padded activation buffers (graph-capture safe)
    (void)hipMemsetAsync(EIN, 0, (size_t)N_EDGES * KPAD_EDGE * sizeof(float), stream);
    (void)hipMemsetAsync(NIN, 0, (size_t)N_NODES * KPAD_NODE * sizeof(float), stream);

    // pack padded inputs
    cpyrows(x,      64, NIN,      KPAD_NODE, N_NODES, 64);
    cpyrows(lap_pe, 16, NIN + 64, KPAD_NODE, N_NODES, 16);
    cpyrows(edge_attr, 16, EIN, KPAD_EDGE, N_EDGES, 16);

    // ---- convert all GEMM weights to bf16 transposed [N,KP] ----
    const bf16_t* WtN0 = convw(P(119), 80, HID, KPAD_NODE);
    const bf16_t* WtN1 = convw(P(121), HID, HID, HID);
    const bf16_t* WtN2 = convw(P(123), HID, HID, HID);
    const bf16_t* WtE0 = convw(P(5), 16, HID, KPAD_EDGE);
    const bf16_t* WtE1 = convw(P(7), HID, HID, HID);
    const bf16_t* WtE2 = convw(P(9), HID, HID, HID);
    const bf16_t *Lgl[4], *Lg0[4], *Lg1[4], *Lg2[4];
    const bf16_t *Lwq[4], *Lwk[4], *Lwv[4], *Lwo[4], *Lm0[4], *Lm1[4];
    for (int L = 0; L < 4; ++L) {
        const int b = 11 + L * 27;
        Lwk[L] = convw(P(b+4),  HID, HID, HID);
        Lwo[L] = convw(P(b+5),  HID, HID, HID);
        Lwq[L] = convw(P(b+6),  HID, HID, HID);
        Lwv[L] = convw(P(b+7),  HID, HID, HID);
        Lgl[L] = convw(P(b+9),  HID, HID, HID);
        Lg0[L] = convw(P(b+11), HID, HID, HID);
        Lg1[L] = convw(P(b+13), HID, HID, HID);
        Lg2[L] = convw(P(b+15), HID, HID, HID);
        Lm0[L] = convw(P(b+23), HID, 2 * HID, HID);
        Lm1[L] = convw(P(b+25), 2 * HID, HID, 2 * HID);
    }
    const bf16_t* WtP0 = convw(P(125), HID, 2 * HID, HID);
    const bf16_t* WtP1 = convw(P(127), 2 * HID, HID, 2 * HID);

    // node_mlp: gelu,gelu,linear
    gemm(NIN, WtN0, P(120), TA, N_NODES, HID, KPAD_NODE, 1);
    gemm(TA,  WtN1, P(122), TB, N_NODES, HID, HID, 1);
    gemm(TB,  WtN2, P(124), H,  N_NODES, HID, HID, 0);

    // edge_mlp: gelu,gelu,linear -> EA
    gemm(EIN,  WtE0, P(6),  EA,   N_EDGES, HID, KPAD_EDGE, 1);
    gemm(EA,   WtE1, P(8),  EBUF, N_EDGES, HID, HID, 1);
    gemm(EBUF, WtE2, P(10), EA,   N_EDGES, HID, HID, 0);

    for (int L = 0; L < 4; ++L) {
        const int b = 11 + L * 27;
        const float *bk = P(b+0), *bo = P(b+1), *bq = P(b+2), *bv = P(b+3);
        const float *eps = P(b+8);
        const float *glB = P(b+10);
        const float *gB0 = P(b+12), *gB1 = P(b+14), *gB2 = P(b+16);
        const float *l1g = P(b+17), *l1b = P(b+18), *l2g = P(b+19),
                    *l2b = P(b+20), *l3g = P(b+21), *l3b = P(b+22);
        const float *mB0 = P(b+24), *mB1 = P(b+26);

        // ---- GINE local branch ----
        gemm(EA, Lgl[L], glB, EBUF, N_EDGES, HID, HID, 0);   // e_lin per layer
        (void)hipMemsetAsync(AGGR, 0, (size_t)nEl * sizeof(float), stream);
        gine_scatter_kernel<<<N_EDGES, HID, 0, stream>>>(H, EBUF, ei, AGGR);
        gine_combine_kernel<<<(nEl + 255) / 256, 256, 0, stream>>>(H, AGGR, eps, TA, nEl);
        gemm(TA, Lg0[L], gB0, TB, N_NODES, HID, HID, 1);
        gemm(TB, Lg1[L], gB1, TC, N_NODES, HID, HID, 1);
        gemm(TC, Lg2[L], gB2, TA, N_NODES, HID, HID, 0);
        addln(TA, H, l1g, l1b, HLOC);                        // LN1(gine + x)

        // ---- global attention branch ----
        gemm(H, Lwq[L], bq, QB, N_NODES, HID, HID, 0);
        gemm(H, Lwk[L], bk, KB, N_NODES, HID, HID, 0);
        gemm(H, Lwv[L], bv, VB, N_NODES, HID, HID, 0);
        attn_kernel<<<(N_NODES * HEADS + 3) / 4, 128, 0, stream>>>(QB, KB, VB, AOUT);
        gemm(AOUT, Lwo[L], bo, TB, N_NODES, HID, HID, 0);
        addln(TB, H, l2g, l2b, HATT);                        // LN2(attn + x)

        // ---- combine + FFN + LN3 + residual ----
        addv(HLOC, HATT, TA);                                // out
        gemm(TA, Lm0[L], mB0, FFNH, N_NODES, 2 * HID, HID, 1);
        gemm(FFNH, Lm1[L], mB1, TB, N_NODES, HID, 2 * HID, 0);
        addv(TA, TB, TC);                                    // out + mlp(out)
        addln(TC, nullptr, l3g, l3b, TB);                    // LN3
        addv(H, TB, H);                                      // h = h + layer(h)
    }

    // postnet + residual
    gemm(H, WtP0, P(126), FFNH, N_NODES, 2 * HID, HID, 1);
    gemm(FFNH, WtP1, P(128), TA, N_NODES, HID, 2 * HID, 0);
    addv(H, TA, H);

    // mean pool + readout
    pool_readout_kernel<<<N_GRAPHS, HID, 0, stream>>>(
        H, P(129), P(130), P(131), P(132), (float*)d_out);
}
